// MultiHeadAttention_1408749273334
// MI455X (gfx1250) — matrix operand order repaired and checked by reading, not compiled, not run on hardware
//
#include <hip/hip_runtime.h>

// ---------------------------------------------------------------------------
// Multi-head attention forward for MI455X (gfx1250, wave32, WMMA + TDM).
//
// B=4, S=2048, D=1024, H=16, dk=64.  ~137 GFLOP, ~150MB compulsory traffic
// -> compute bound; all GEMMs on v_wmma_f32_16x16x32_bf16, f32 accumulate.
// GEMM k-loops unrolled by 2 so the double-buffered fragments swap roles
// without register copies (removes v_dual_mov churn + WMMA hazard nops).
// Attention double-buffers K/V LDS tiles: chunk i+1's TDM (or cooperative
// copy) overlaps chunk i's WMMAs and softmax.
//
// Workspace layout (72 MB):
//   [ 0.. 8MB)  Wq/Wk/Wv/Wo bf16, 2MB each
//   [ 8..24MB)  Qb  bf16 [B*H][S][64]
//   [24..40MB)  Kb  bf16 [B*H][S][64]
//   [40..56MB)  Vt  bf16 [B*H][64][S]   (transposed for PV B-fragments)
//   [56..72MB)  Ctx bf16 [B][S][D]
// ---------------------------------------------------------------------------

#define DMODEL 1024
#define SEQ    2048
#define NHEAD  16
#define DK     64
#define BATCH  4

typedef unsigned short u16;
typedef unsigned int   u32;
typedef __attribute__((ext_vector_type(16))) __bf16 v16bf;
typedef __attribute__((ext_vector_type(8)))  __bf16 v8bf;
typedef __attribute__((ext_vector_type(8)))  float  v8f;
typedef __attribute__((ext_vector_type(4)))  unsigned int u32x4;
typedef __attribute__((ext_vector_type(8)))  int    i32x8;
typedef __attribute__((ext_vector_type(4)))  int    i32x4;

#if __has_builtin(__builtin_amdgcn_tensor_load_to_lds)
#define HAVE_TDM 1
#endif

// ---------------------------- fragment helpers -----------------------------

static __device__ __forceinline__ v8bf ld8bf(const u16* base, int off) {
    uint4 u = *reinterpret_cast<const uint4*>(base + off);
    return __builtin_bit_cast(v8bf, u);
}

static __device__ __forceinline__ v16bf cat16(v8bf lo, v8bf hi) {
    v16bf r;
#pragma unroll
    for (int i = 0; i < 8; i++) { r[i] = lo[i]; r[i + 8] = hi[i]; }
    return r;
}

// A-fragment 16x32 from a bf16 row: k = off+{0..7} U off+16+{0..7}
static __device__ __forceinline__ v16bf fragA_bf16(const u16* row, int off) {
    return cat16(ld8bf(row, off), ld8bf(row, off + 16));
}

// B-fragment 32x16 from a bf16 row: k = off+{0..7} U off+16+{0..7}
static __device__ __forceinline__ v16bf fragB_bf16(const u16* row, int off) {
    return cat16(ld8bf(row, off), ld8bf(row, off + 16));
}

// A-fragment from an fp32 row: issue all 4 b128 loads, then convert.
static __device__ __forceinline__ v16bf fragA_f32(const float* row, int off) {
    const float4* p0 = reinterpret_cast<const float4*>(row + off);
    const float4* p1 = reinterpret_cast<const float4*>(row + off + 16);
    float4 f0 = p0[0], f1 = p0[1], f2 = p1[0], f3 = p1[1];
    v16bf r;
    r[0]  = (__bf16)f0.x; r[1]  = (__bf16)f0.y; r[2]  = (__bf16)f0.z; r[3]  = (__bf16)f0.w;
    r[4]  = (__bf16)f1.x; r[5]  = (__bf16)f1.y; r[6]  = (__bf16)f1.z; r[7]  = (__bf16)f1.w;
    r[8]  = (__bf16)f2.x; r[9]  = (__bf16)f2.y; r[10] = (__bf16)f2.z; r[11] = (__bf16)f2.w;
    r[12] = (__bf16)f3.x; r[13] = (__bf16)f3.y; r[14] = (__bf16)f3.z; r[15] = (__bf16)f3.w;
    return r;
}

static __device__ __forceinline__ u16 f2bf(float f) {
    return __builtin_bit_cast(u16, (__bf16)f);
}

static __device__ __forceinline__ v8f wmma_bf16(v16bf a, v16bf b, v8f c) {
    return __builtin_amdgcn_wmma_f32_16x16x32_bf16(false, a, false, b, (short)0, c,
                                                   false, false);
}

// ---------------------------- TDM staging ----------------------------------
#if defined(HAVE_TDM)
static __device__ __forceinline__ u32 lds_off(const void* p) {
    // LDS offset for TDM descriptor (flat addr low bits; compile-time probe env)
    return (u32)(unsigned long long)p;
}

// 2D tile load: tile1 rows x tile0 elements (bf16), global row stride stride0.
static __device__ __forceinline__ void tdm_load_2d(const void* gptr, u32 lds,
                                                   u32 tile0, u32 tile1,
                                                   u32 td0, u32 td1, u32 stride0)
{
    unsigned long long ga = (unsigned long long)gptr;
    u32x4 g0;
    g0[0] = 1u;                                       // count=1, user descriptor
    g0[1] = lds;                                      // lds_addr
    g0[2] = (u32)ga;                                  // global_addr[95:64]
    g0[3] = ((u32)(ga >> 32) & 0x1FFFFFFu) | (2u << 30);  // addr[120:96] | type=2
    i32x8 g1;
    g1[0] = (int)(1u << 16);                          // data_size=1 (2 bytes)
    g1[1] = (int)((td0 & 0xFFFFu) << 16);             // tensor_dim0[15:0]
    g1[2] = (int)((td0 >> 16) | ((td1 & 0xFFFFu) << 16));
    g1[3] = (int)((td1 >> 16) | (tile0 << 16));       // tile_dim0
    g1[4] = (int)tile1;                               // tile_dim1 (tile_dim2=0)
    g1[5] = (int)stride0;                             // tensor_dim0_stride[31:0]
    g1[6] = 0;
    g1[7] = 0;
    i32x4 z4 = {0, 0, 0, 0};
#if __clang_major__ >= 23
    i32x8 z8 = {0, 0, 0, 0, 0, 0, 0, 0};
    __builtin_amdgcn_tensor_load_to_lds(g0, g1, z4, z4, z8, 0);
#else
    __builtin_amdgcn_tensor_load_to_lds(g0, g1, z4, z4, 0);
#endif
}
#endif

// Stage one 64-key chunk of K ([key][dk]) and V ([dk][key]) into LDS.
static __device__ __forceinline__ void stage_tiles(const u16* __restrict__ Kb,
                                                   const u16* __restrict__ Vb,
                                                   int key0, u16* kt, u16* vt,
                                                   int tid, int wv)
{
#if defined(HAVE_TDM)
    if (wv == 0) {
        tdm_load_2d(Kb + (size_t)key0 * DK, lds_off(kt), DK, 64, DK, SEQ - key0, DK);
        tdm_load_2d(Vb + key0, lds_off(vt), 64, DK, SEQ - key0, DK, SEQ);
    }
#else
    const uint4* gk = reinterpret_cast<const uint4*>(Kb + (size_t)key0 * DK);
    uint4* lk = reinterpret_cast<uint4*>(kt);
    for (int i = tid; i < 512; i += 128) lk[i] = gk[i];
    uint4* lv = reinterpret_cast<uint4*>(vt);
    for (int i = tid; i < 512; i += 128) {
        const int row = i >> 3, colc = i & 7;
        lv[i] = reinterpret_cast<const uint4*>(Vb + (size_t)row * SEQ + key0)[colc];
    }
#endif
}

// ---------------------------------------------------------------------------
// Weight conversion: fp32 -> bf16, once per matrix.  n4 = elements / 4.
// ---------------------------------------------------------------------------
__global__ __launch_bounds__(256)
void cvt_w(const float* __restrict__ src, u16* __restrict__ dst, int n4)
{
    const int i = blockIdx.x * 256 + threadIdx.x;
    if (i < n4) {
        float4 f = reinterpret_cast<const float4*>(src)[i];
        uint2 p;
        p.x = (u32)f2bf(f.x) | ((u32)f2bf(f.y) << 16);
        p.y = (u32)f2bf(f.z) | ((u32)f2bf(f.w) << 16);
        reinterpret_cast<uint2*>(dst)[i] = p;
    }
}

// ---------------------------------------------------------------------------
// Input projection: Y = X @ W^T (X fp32 [8192,1024], W bf16 [1024,1024]).
// Block = 8 waves, tile 128(M) x 64(N); wave: 16x64.
// k-loop unrolled by 2: buffers 0/1 alternate producer/consumer (no copies).
// ---------------------------------------------------------------------------
__global__ __launch_bounds__(256)
void mha_proj_in(const float* __restrict__ x, const u16* __restrict__ w,
                 u16* __restrict__ out, int transpose_v)
{
    const int tid  = threadIdx.x;
    const int lane = tid & 31, wv = tid >> 5;
    const int g = lane >> 4, l15 = lane & 15;
    const int m0 = blockIdx.x * 128 + wv * 16;
    const int n0 = blockIdx.y * 64;

    v8f c[4];
#pragma unroll
    for (int t = 0; t < 4; t++)
#pragma unroll
        for (int i = 0; i < 8; i++) c[t][i] = 0.0f;

    const float* xrow  = x + (size_t)(m0 + l15) * DMODEL;
    const u16*   wbase = w + (size_t)(n0 + l15) * DMODEL;

    v16bf a0 = fragA_f32(xrow, g * 8);
    v16bf b0[4], a1, b1[4];
#pragma unroll
    for (int nt = 0; nt < 4; nt++)
        b0[nt] = fragB_bf16(wbase, nt * 16 * DMODEL + g * 8);

    int k = 0;
#pragma unroll 1
    for (; k + 64 < DMODEL; k += 64) {
        __builtin_prefetch(xrow + k + 128, 0, 1);
        a1 = fragA_f32(xrow, k + 32 + g * 8);
#pragma unroll
        for (int nt = 0; nt < 4; nt++)
            b1[nt] = fragB_bf16(wbase, nt * 16 * DMODEL + k + 32 + g * 8);
#pragma unroll
        for (int nt = 0; nt < 4; nt++)
            c[nt] = wmma_bf16(a0, b0[nt], c[nt]);
        a0 = fragA_f32(xrow, k + 64 + g * 8);
#pragma unroll
        for (int nt = 0; nt < 4; nt++)
            b0[nt] = fragB_bf16(wbase, nt * 16 * DMODEL + k + 64 + g * 8);
#pragma unroll
        for (int nt = 0; nt < 4; nt++)
            c[nt] = wmma_bf16(a1, b1[nt], c[nt]);
    }
    // tail: buffers hold k = DMODEL-64; last step at k = DMODEL-32
    a1 = fragA_f32(xrow, DMODEL - 32 + g * 8);
#pragma unroll
    for (int nt = 0; nt < 4; nt++)
        b1[nt] = fragB_bf16(wbase, nt * 16 * DMODEL + DMODEL - 32 + g * 8);
#pragma unroll
    for (int nt = 0; nt < 4; nt++)
        c[nt] = wmma_bf16(a0, b0[nt], c[nt]);
#pragma unroll
    for (int nt = 0; nt < 4; nt++)
        c[nt] = wmma_bf16(a1, b1[nt], c[nt]);

#pragma unroll
    for (int nt = 0; nt < 4; nt++) {
        const int n = n0 + nt * 16 + l15;
        const int h = n >> 6, dd = n & 63;
#pragma unroll
        for (int r = 0; r < 8; r++) {
            const int m  = m0 + r + 8 * g;           // C/D: m = vgpr + 8*(lane/16)
            const int bb = m >> 11, s = m & (SEQ - 1);
            const int bh = bb * NHEAD + h;
            const size_t idx = transpose_v
                ? ((size_t)bh * DK + dd) * SEQ + s
                : ((size_t)bh * SEQ + s) * DK + dd;
            out[idx] = f2bf(c[nt][r]);
        }
    }
}

// ---------------------------------------------------------------------------
// Flash attention: block = 4 waves, 64 q-rows of one (b,h).  K/V tiles are
// double-buffered in LDS; chunk i+1's TDM overlaps chunk i's compute.
// ---------------------------------------------------------------------------
__global__ __launch_bounds__(128)
void mha_attn(const u16* __restrict__ Q, const u16* __restrict__ K,
              const u16* __restrict__ Vt, u16* __restrict__ ctx)
{
    __shared__ u16 ktile[2][64 * 64];                  // [key][dk]   2 x 8 KB
    __shared__ u16 vtile[2][64 * 64];                  // [dk][key]   2 x 8 KB
    __shared__ u16 pbuf[4 * 16 * 64];                  // per-wave P      8 KB
    const int tid  = threadIdx.x;
    const int lane = tid & 31, wv = tid >> 5;
    const int g = lane >> 4, l15 = lane & 15;
    const int bh = blockIdx.y;
    const int q0 = blockIdx.x * 64 + wv * 16;
    const u16* Qb = Q  + (size_t)bh * SEQ * DK;
    const u16* Kb = K  + (size_t)bh * SEQ * DK;
    const u16* Vb = Vt + (size_t)bh * DK * SEQ;
    u16* pb = pbuf + wv * 16 * 64;

    // Loop-invariant Q A-fragments (dk=64 -> 2 k-steps of 32).
    v16bf aq[2];
    const u16* qrow = Qb + (size_t)(q0 + l15) * DK;
#pragma unroll
    for (int st = 0; st < 2; st++)
        aq[st] = fragA_bf16(qrow, st * 32 + g * 8);

    float mrun[8], lrun[8];
    v8f acc[4];
#pragma unroll
    for (int r = 0; r < 8; r++) { mrun[r] = -3.0e38f; lrun[r] = 0.0f; }
#pragma unroll
    for (int t = 0; t < 4; t++)
#pragma unroll
        for (int i = 0; i < 8; i++) acc[t][i] = 0.0f;

    stage_tiles(Kb, Vb, 0, ktile[0], vtile[0], tid, wv);

    for (int key0 = 0; key0 < SEQ; key0 += 64) {
        const int buf = (key0 >> 6) & 1;
#if defined(HAVE_TDM)
        if (wv == 0) __builtin_amdgcn_s_wait_tensorcnt(0);   // tile[buf] arrived
#endif
        __syncthreads();   // publish tile[buf]; prior reads of tile[buf^1] done
        if (key0 + 64 < SEQ)
            stage_tiles(Kb, Vb, key0 + 64, ktile[buf ^ 1], vtile[buf ^ 1], tid, wv);
        const u16* kt = ktile[buf];
        const u16* vt = vtile[buf];

        // ---- scores: S = Q K^T from LDS K tile ----------------------------
        v8f sc[4];
#pragma unroll
        for (int t = 0; t < 4; t++)
#pragma unroll
            for (int i = 0; i < 8; i++) sc[t][i] = 0.0f;
#pragma unroll
        for (int st = 0; st < 2; st++)
#pragma unroll
            for (int t = 0; t < 4; t++) {
                v16bf bk = fragB_bf16(kt, (t * 16 + l15) * 64 + st * 32 + g * 8);
                sc[t] = wmma_bf16(aq[st], bk, sc[t]);
            }

        // ---- online softmax (row m = r + 8*g lives in lane halves) --------
#pragma unroll
        for (int r = 0; r < 8; r++) {
            float v = -3.0e38f;
#pragma unroll
            for (int t = 0; t < 4; t++) { sc[t][r] *= 0.125f; v = fmaxf(v, sc[t][r]); }
#pragma unroll
            for (int msk = 1; msk < 16; msk <<= 1) v = fmaxf(v, __shfl_xor(v, msk, 32));
            const float mn    = fmaxf(mrun[r], v);
            const float alpha = __expf(mrun[r] - mn);
            mrun[r] = mn;
            float ps = 0.0f;
#pragma unroll
            for (int t = 0; t < 4; t++) {
                const float p = __expf(sc[t][r] - mn);
                sc[t][r] = p; ps += p;
            }
#pragma unroll
            for (int msk = 1; msk < 16; msk <<= 1) ps += __shfl_xor(ps, msk, 32);
            lrun[r] = lrun[r] * alpha + ps;
#pragma unroll
            for (int t = 0; t < 4; t++) acc[t][r] *= alpha;
        }

        // ---- P: C/D layout -> LDS -> A-fragment layout --------------------
#pragma unroll
        for (int t = 0; t < 4; t++)
#pragma unroll
            for (int r = 0; r < 8; r++)
                pb[(r + 8 * g) * 64 + t * 16 + l15] = f2bf(sc[t][r]);
        __builtin_amdgcn_wave_barrier();               // same-wave DS in-order
        v16bf ap[2];
        const u16* prow = pb + l15 * 64;
#pragma unroll
        for (int st = 0; st < 2; st++)
            ap[st] = fragA_bf16(prow, st * 32 + g * 8);

        // ---- acc += P @ V from LDS V tile ---------------------------------
#pragma unroll
        for (int st = 0; st < 2; st++)
#pragma unroll
            for (int dt = 0; dt < 4; dt++) {
                v16bf bv = fragB_bf16(vt, (dt * 16 + l15) * 64 + st * 32 + g * 8);
                acc[dt] = wmma_bf16(ap[st], bv, acc[dt]);
            }
    }

    // ---- epilogue: normalize, write ctx[b][s][h*64+d] ---------------------
    const int bb = bh >> 4, h = bh & (NHEAD - 1);
#pragma unroll
    for (int r = 0; r < 8; r++) {
        const float inv = 1.0f / lrun[r];
        const int srow = q0 + r + 8 * g;
#pragma unroll
        for (int dt = 0; dt < 4; dt++) {
            const int n = h * DK + dt * 16 + l15;
            ctx[((size_t)bb * SEQ + srow) * DMODEL + n] = f2bf(acc[dt][r] * inv);
        }
    }
}

// ---------------------------------------------------------------------------
// Output projection: out = Ctx @ Wo^T + b_o  (Ctx bf16, Wo bf16, out fp32)
// ---------------------------------------------------------------------------
__global__ __launch_bounds__(256)
void mha_proj_out(const u16* __restrict__ ctx, const u16* __restrict__ w,
                  const float* __restrict__ bias, float* __restrict__ out)
{
    const int tid  = threadIdx.x;
    const int lane = tid & 31, wv = tid >> 5;
    const int g = lane >> 4, l15 = lane & 15;
    const int m0 = blockIdx.x * 128 + wv * 16;
    const int n0 = blockIdx.y * 64;

    v8f c[4];
#pragma unroll
    for (int t = 0; t < 4; t++)
#pragma unroll
        for (int i = 0; i < 8; i++) c[t][i] = 0.0f;

    const u16* crow  = ctx + (size_t)(m0 + l15) * DMODEL;
    const u16* wbase = w + (size_t)(n0 + l15) * DMODEL;

    v16bf a0 = fragA_bf16(crow, g * 8);
    v16bf b0[4], a1, b1[4];
#pragma unroll
    for (int nt = 0; nt < 4; nt++)
        b0[nt] = fragB_bf16(wbase, nt * 16 * DMODEL + g * 8);

    int k = 0;
#pragma unroll 1
    for (; k + 64 < DMODEL; k += 64) {
        __builtin_prefetch(crow + k + 128, 0, 1);
        a1 = fragA_bf16(crow, k + 32 + g * 8);
#pragma unroll
        for (int nt = 0; nt < 4; nt++)
            b1[nt] = fragB_bf16(wbase, nt * 16 * DMODEL + k + 32 + g * 8);
#pragma unroll
        for (int nt = 0; nt < 4; nt++)
            c[nt] = wmma_bf16(a0, b0[nt], c[nt]);
        a0 = fragA_bf16(crow, k + 64 + g * 8);
#pragma unroll
        for (int nt = 0; nt < 4; nt++)
            b0[nt] = fragB_bf16(wbase, nt * 16 * DMODEL + k + 64 + g * 8);
#pragma unroll
        for (int nt = 0; nt < 4; nt++)
            c[nt] = wmma_bf16(a1, b1[nt], c[nt]);
    }
    a1 = fragA_bf16(crow, DMODEL - 32 + g * 8);
#pragma unroll
    for (int nt = 0; nt < 4; nt++)
        b1[nt] = fragB_bf16(wbase, nt * 16 * DMODEL + DMODEL - 32 + g * 8);
#pragma unroll
    for (int nt = 0; nt < 4; nt++)
        c[nt] = wmma_bf16(a0, b0[nt], c[nt]);
#pragma unroll
    for (int nt = 0; nt < 4; nt++)
        c[nt] = wmma_bf16(a1, b1[nt], c[nt]);

#pragma unroll
    for (int nt = 0; nt < 4; nt++) {
        const int n  = n0 + nt * 16 + l15;
        const float bo = bias[n];
#pragma unroll
        for (int r = 0; r < 8; r++) {
            const int m = m0 + r + 8 * g;
            out[(size_t)m * DMODEL + n] = c[nt][r] + bo;
        }
    }
}

// ---------------------------------------------------------------------------
extern "C" void kernel_launch(void* const* d_in, const int* in_sizes, int n_in,
                              void* d_out, int out_size, void* d_ws, size_t ws_size,
                              hipStream_t stream)
{
    const float* query = (const float*)d_in[0];
    const float* key_  = (const float*)d_in[1];
    const float* value = (const float*)d_in[2];
    const float* w_q   = (const float*)d_in[3];
    const float* w_k   = (const float*)d_in[4];
    const float* w_v   = (const float*)d_in[5];
    const float* w_o   = (const float*)d_in[6];
    const float* b_o   = (const float*)d_in[7];
    float* out = (float*)d_out;

    char* ws = (char*)d_ws;
    const size_t MB = (size_t)1024 * 1024;
    u16* Wq  = (u16*)(ws + 0 * MB);
    u16* Wk  = (u16*)(ws + 2 * MB);
    u16* Wv  = (u16*)(ws + 4 * MB);
    u16* Wo  = (u16*)(ws + 6 * MB);
    u16* Qb  = (u16*)(ws + 8 * MB);
    u16* Kb  = (u16*)(ws + 24 * MB);
    u16* Vt  = (u16*)(ws + 40 * MB);
    u16* Ctx = (u16*)(ws + 56 * MB);

    // 1) weights fp32 -> bf16 (once)
    const int n4 = DMODEL * DMODEL / 4;
    const dim3 gc((n4 + 255) / 256), bc(256);
    cvt_w<<<gc, bc, 0, stream>>>(w_q, Wq, n4);
    cvt_w<<<gc, bc, 0, stream>>>(w_k, Wk, n4);
    cvt_w<<<gc, bc, 0, stream>>>(w_v, Wv, n4);
    cvt_w<<<gc, bc, 0, stream>>>(w_o, Wo, n4);

    // 2) projections
    const dim3 gp(64, 16), bp(256);   // (M=8192)/128 x (N=1024)/64
    mha_proj_in<<<gp, bp, 0, stream>>>(query, Wq, Qb, 0);
    mha_proj_in<<<gp, bp, 0, stream>>>(key_,  Wk, Kb, 0);
    mha_proj_in<<<gp, bp, 0, stream>>>(value, Wv, Vt, 1);

    // 3) attention
    mha_attn<<<dim3(SEQ / 64, BATCH * NHEAD), dim3(128), 0, stream>>>(Qb, Kb, Vt, Ctx);

    // 4) output projection + bias
    mha_proj_out<<<gp, bp, 0, stream>>>(Ctx, Wo, b_o, out);
}
